// QJoint_45002667327553
// MI455X (gfx1250) — matrix operand order repaired and checked by reading, not compiled, not run on hardware
//
#include <hip/hip_runtime.h>

typedef __attribute__((ext_vector_type(16))) _Float16 v16h;
typedef __attribute__((ext_vector_type(8)))  _Float16 v8h;
typedef __attribute__((ext_vector_type(8)))  float    v8f;
typedef int v4i_t __attribute__((vector_size(16)));

__device__ __forceinline__ float eluf(float x) { return x > 0.f ? x : __expf(x) - 1.f; }

// A-matrix fragment (16x32 f16): lane<16 -> M=lane, K={0..7,16..23}; lane>=16 -> M=lane-16, K={8..15,24..31}
__device__ __forceinline__ v16h frag_a(const _Float16* row, int kchunk, int lane) {
    const int off = kchunk * 32 + ((lane & 16) ? 8 : 0);
    v8h lo = *(const v8h*)(row + off);
    v8h hi = *(const v8h*)(row + off + 16);
    return __builtin_shufflevector(lo, hi, 0,1,2,3,4,5,6,7,8,9,10,11,12,13,14,15);
}

// B-matrix fragment (32x16 f16), weights stored [N][K]: lane<16 -> N=lane, K={0..7,16..23}; lane>=16 -> N=lane-16, K={8..15,24..31}
__device__ __forceinline__ v16h frag_b(const _Float16* wrow, int kchunk, int lane) {
    const int off = kchunk * 32 + ((lane & 16) ? 8 : 0);
    v8h lo = *(const v8h*)(wrow + off);
    v8h hi = *(const v8h*)(wrow + off + 16);
    return __builtin_shufflevector(lo, hi, 0,1,2,3,4,5,6,7,8,9,10,11,12,13,14,15);
}

#define WMMA_F16(a, b, c) \
    __builtin_amdgcn_wmma_f32_16x16x32_f16(false, (a), false, (b), (short)0, (c), false, false)

// ---- CDNA5 async global->LDS copy (ASYNCcnt path), compile-safe fallback ----
#if __has_builtin(__builtin_amdgcn_global_load_async_to_lds_b128)
#define HAVE_ASYNC_LDS 1
__device__ __forceinline__ void async_copy16(const float* g, float* l) {
    // param types per hipcc diagnostic: (int4 __device__*, int4 __shared__*, Imm, Imm);
    // generic pointers implicitly convert to CUDA-AS pointers in HIP.
    __builtin_amdgcn_global_load_async_to_lds_b128((v4i_t*)g, (v4i_t*)l, 0, 0);
}
#else
#define HAVE_ASYNC_LDS 0
__device__ __forceinline__ void async_copy16(const float* g, float* l) {
    *(float4*)l = *(const float4*)g;
}
#endif

__device__ __forceinline__ void async_wait0() {
#if HAVE_ASYNC_LDS
#if __has_builtin(__builtin_amdgcn_s_wait_asynccnt)
    __builtin_amdgcn_s_wait_asynccnt(0);
#else
    asm volatile("s_wait_asynccnt 0x0" ::: "memory");
#endif
#endif
}

// ---------------------------------------------------------------------------
// Kernel 1: key1 = MLP0(action)  [P,66] -> [P,64], 6 layers, ELU between
// ---------------------------------------------------------------------------
__global__ __launch_bounds__(256)
void mlp0_kernel(const float* __restrict__ X,
                 const float* __restrict__ w0p, const float* __restrict__ w1p,
                 const float* __restrict__ w2p, const float* __restrict__ w3p,
                 const float* __restrict__ w4p, const float* __restrict__ w5p,
                 const float* __restrict__ b0p, const float* __restrict__ b1p,
                 const float* __restrict__ b2p, const float* __restrict__ b3p,
                 const float* __restrict__ b4p, const float* __restrict__ b5p,
                 float* __restrict__ key1, int P)
{
    __shared__ _Float16 sW0[64 * 96];      // layer0 weights, [n][k], K padded 66->96
    __shared__ _Float16 sW[5][64 * 64];    // layers 1..5, [n][k]
    __shared__ float    sB[6][64];
    __shared__ _Float16 sX[128 * 96];      // layer0 input (stride 96), reused (stride 64) for odd-layer outputs
    __shared__ _Float16 sA[128 * 64];      // even-layer outputs

    const int t = threadIdx.x;
    // ---- stage weights (transpose to [n][k], convert f32->f16) ----
    for (int j = t; j < 64 * 96; j += 256) {
        int n = j / 96, k = j % 96;
        sW0[j] = (k < 66) ? (_Float16)w0p[k * 64 + n] : (_Float16)0.f;
    }
    const float* wptr[5] = { w1p, w2p, w3p, w4p, w5p };
#pragma unroll
    for (int l = 0; l < 5; ++l)
        for (int j = t; j < 64 * 64; j += 256) {
            int n = j >> 6, k = j & 63;
            sW[l][j] = (_Float16)wptr[l][k * 64 + n];
        }
    if (t < 64) {
        sB[0][t] = b0p[t]; sB[1][t] = b1p[t]; sB[2][t] = b2p[t];
        sB[3][t] = b3p[t]; sB[4][t] = b4p[t]; sB[5][t] = b5p[t];
    }
    // ---- stage 128 input rows, pad K 66->96 with zeros ----
    const size_t row0 = (size_t)blockIdx.x * 128;
    for (int j = t; j < 128 * 96; j += 256) {
        int r = j / 96, k = j % 96;
        size_t rr = row0 + r; if (rr >= (size_t)P) rr = (size_t)P - 1;
        sX[j] = (k < 66) ? (_Float16)X[rr * 66 + k] : (_Float16)0.f;
    }
    __syncthreads();

    const int lane = t & 31, wv = t >> 5;
    const int m = lane & 15;
    const int mhi = (lane & 16) ? 8 : 0;

    // ---- layer 0 : K=96 (3 chunks), out -> sA ----
    {
        const _Float16* arow = sX + (size_t)(wv * 16 + m) * 96;
        v16h a0 = frag_a(arow, 0, lane);
        v16h a1 = frag_a(arow, 1, lane);
        v16h a2 = frag_a(arow, 2, lane);
#pragma unroll
        for (int nt = 0; nt < 4; ++nt) {
            const int n = nt * 16 + (lane & 15);
            const float bv = sB[0][n];
            v8f acc = { bv, bv, bv, bv, bv, bv, bv, bv };
            acc = WMMA_F16(a0, frag_b(sW0 + (size_t)n * 96, 0, lane), acc);
            acc = WMMA_F16(a1, frag_b(sW0 + (size_t)n * 96, 1, lane), acc);
            acc = WMMA_F16(a2, frag_b(sW0 + (size_t)n * 96, 2, lane), acc);
#pragma unroll
            for (int r = 0; r < 8; ++r)
                sA[(size_t)(wv * 16 + r + mhi) * 64 + n] = (_Float16)eluf(acc[r]);
        }
        asm volatile("s_wait_dscnt 0x0" ::: "memory");
    }

    // ---- layers 1..5 : K=64 (2 chunks) ----
#pragma unroll
    for (int layer = 1; layer < 6; ++layer) {
        const _Float16* W    = sW[layer - 1];
        const _Float16* inb  = (layer & 1) ? sA : sX;
        _Float16*       outb = (layer & 1) ? sX : sA;
        const _Float16* arow = inb + (size_t)(wv * 16 + m) * 64;
        v16h a0 = frag_a(arow, 0, lane);
        v16h a1 = frag_a(arow, 1, lane);
#pragma unroll
        for (int nt = 0; nt < 4; ++nt) {
            const int n = nt * 16 + (lane & 15);
            const float bv = sB[layer][n];
            v8f acc = { bv, bv, bv, bv, bv, bv, bv, bv };
            acc = WMMA_F16(a0, frag_b(W + (size_t)n * 64, 0, lane), acc);
            acc = WMMA_F16(a1, frag_b(W + (size_t)n * 64, 1, lane), acc);
            if (layer < 5) {
#pragma unroll
                for (int r = 0; r < 8; ++r)
                    outb[(size_t)(wv * 16 + r + mhi) * 64 + n] = (_Float16)eluf(acc[r]);
            } else {
#pragma unroll
                for (int r = 0; r < 8; ++r) {
                    size_t grow = row0 + (size_t)(wv * 16 + r + mhi);
                    if (grow < (size_t)P) key1[grow * 64 + n] = acc[r];
                }
            }
        }
        asm volatile("s_wait_dscnt 0x0" ::: "memory");
    }
}

// ---------------------------------------------------------------------------
// Kernel 2: segment sums + counts (member_group sorted -> LDS window atomics)
// ---------------------------------------------------------------------------
#define SLOTS 64
__global__ __launch_bounds__(256)
void segsum_kernel(const float* __restrict__ key1, const int* __restrict__ mp,
                   const int* __restrict__ mg, float* __restrict__ sums,
                   float* __restrict__ cnts, int M, int G)
{
    __shared__ float sAcc[SLOTS * 64];
    __shared__ float sCnt[SLOTS];
    __shared__ int   sPair[256];
    __shared__ int   sGrp[256];

    const int t = threadIdx.x;
    const size_t e0 = (size_t)blockIdx.x * 256;
    for (int j = t; j < SLOTS * 64; j += 256) sAcc[j] = 0.f;
    if (t < SLOTS) sCnt[t] = 0.f;
    {
        size_t e = e0 + t;
        if (e < (size_t)M) { sPair[t] = mp[e]; sGrp[t] = mg[e]; }
        else               { sPair[t] = -1;    sGrp[t] = -1;    }
    }
    __syncthreads();
    const int gmin = sGrp[0];  // first edge of block is always valid

    // counts: one edge per thread
    {
        int g = sGrp[t];
        if (g >= 0) {
            int slot = g - gmin;
            if (slot >= 0 && slot < SLOTS) atomicAdd(&sCnt[slot], 1.f);
            else                           atomicAdd(&cnts[g], 1.f);
        }
    }
    // sums: thread t owns column k=t&63 over edges e = (t>>6) + 4*i  (coalesced key1 reads)
    for (int j = t; j < 256 * 64; j += 256) {
        int e = j >> 6, k = j & 63;
        int g = sGrp[e];
        if (g < 0) continue;
        float v = key1[(size_t)sPair[e] * 64 + k];
        int slot = g - gmin;
        if (slot >= 0 && slot < SLOTS) atomicAdd(&sAcc[slot * 64 + k], v);
        else                           atomicAdd(&sums[(size_t)g * 64 + k], v);
    }
    __syncthreads();
    // flush LDS window to global
    for (int j = t; j < SLOTS * 64; j += 256) {
        float v = sAcc[j];
        int g = gmin + (j >> 6);
        if (v != 0.f && g < G) atomicAdd(&sums[(size_t)g * 64 + (j & 63)], v);
    }
    if (t < SLOTS) {
        float c = sCnt[t]; int g = gmin + t;
        if (c != 0.f && g < G) atomicAdd(&cnts[g], c);
    }
}

// ---------------------------------------------------------------------------
// Kernel 3: group_mean -> head MLP (64->128->64->1) -> b_qjt[G]
// ---------------------------------------------------------------------------
__global__ __launch_bounds__(256)
void head_kernel(const float* __restrict__ sums, const float* __restrict__ cnts,
                 const float* __restrict__ hw0, const float* __restrict__ hb0,
                 const float* __restrict__ hw1, const float* __restrict__ hb1,
                 const float* __restrict__ hw2, const float* __restrict__ hb2,
                 float* __restrict__ bq, int G)
{
    __shared__ _Float16 sW1[128 * 64];    // [n][k]  64->128
    __shared__ _Float16 sW2[64 * 128];    // [n][k]  128->64
    __shared__ float    sB1[128];
    __shared__ float    sB2[64];
    __shared__ float    sW3[64];
    __shared__ _Float16 sMean[128 * 64];  // input rows; reused as act2
    // overlay: fp32 async-staged sums block (phase 1) vs f16 act1 (phase 2)
    __shared__ union {
        float    sums32[128 * 64];    // 32 KB
        _Float16 act1[128 * 128];     // 32 KB
    } sOv;

    const int t = threadIdx.x;
    for (int j = t; j < 128 * 64; j += 256) { int n = j >> 6, k = j & 63;  sW1[j] = (_Float16)hw0[k * 128 + n]; }
    for (int j = t; j < 64 * 128; j += 256) { int n = j >> 7, k = j & 127; sW2[j] = (_Float16)hw1[k * 64 + n]; }
    if (t < 128) sB1[t] = hb0[t];
    if (t < 64)  { sB2[t] = hb1[t]; sW3[t] = hw2[t]; }

    const size_t g0 = (size_t)blockIdx.x * 128;
    // async bulk-copy of the contiguous 32KB sums slab for this block
#pragma unroll
    for (int i = 0; i < 8; ++i) {
        int c = t + 256 * i;                  // 16B chunk id, 0..2047
        async_copy16(sums + g0 * 64 + (size_t)c * 4, sOv.sums32 + c * 4);
    }
    async_wait0();
    __syncthreads();
    for (int j = t; j < 128 * 64; j += 256) {
        int r = j >> 6;
        size_t g = g0 + r;
        float v = 0.f;
        if (g < (size_t)G) { float c = fmaxf(cnts[g], 1.f); v = sOv.sums32[j] / c; }
        sMean[j] = (_Float16)v;
    }
    __syncthreads();   // also separates sOv.sums32 reads from sOv.act1 writes

    const int lane = t & 31, wv = t >> 5;
    const int m = lane & 15;
    const int mhi = (lane & 16) ? 8 : 0;

    // layer 1: K=64, N=128
    {
        const _Float16* arow = sMean + (size_t)(wv * 16 + m) * 64;
        v16h a0 = frag_a(arow, 0, lane);
        v16h a1 = frag_a(arow, 1, lane);
#pragma unroll
        for (int nt = 0; nt < 8; ++nt) {
            const int n = nt * 16 + (lane & 15);
            const float bv = sB1[n];
            v8f acc = { bv, bv, bv, bv, bv, bv, bv, bv };
            acc = WMMA_F16(a0, frag_b(sW1 + (size_t)n * 64, 0, lane), acc);
            acc = WMMA_F16(a1, frag_b(sW1 + (size_t)n * 64, 1, lane), acc);
#pragma unroll
            for (int r = 0; r < 8; ++r)
                sOv.act1[(size_t)(wv * 16 + r + mhi) * 128 + n] = (_Float16)eluf(acc[r]);
        }
        asm volatile("s_wait_dscnt 0x0" ::: "memory");
    }
    // layer 2: K=128, N=64, output overwrites sMean (per-wave private rows)
    {
        const _Float16* arow = sOv.act1 + (size_t)(wv * 16 + m) * 128;
        v16h af[4];
#pragma unroll
        for (int c = 0; c < 4; ++c) af[c] = frag_a(arow, c, lane);
#pragma unroll
        for (int nt = 0; nt < 4; ++nt) {
            const int n = nt * 16 + (lane & 15);
            const float bv = sB2[n];
            v8f acc = { bv, bv, bv, bv, bv, bv, bv, bv };
#pragma unroll
            for (int c = 0; c < 4; ++c)
                acc = WMMA_F16(af[c], frag_b(sW2 + (size_t)n * 128, c, lane), acc);
#pragma unroll
            for (int r = 0; r < 8; ++r)
                sMean[(size_t)(wv * 16 + r + mhi) * 64 + n] = (_Float16)eluf(acc[r]);
        }
        asm volatile("s_wait_dscnt 0x0" ::: "memory");
    }
    // final: 64 -> 1 (two lanes per row, shuffle combine)
    {
        const int row = lane & 15, hsel = lane >> 4;
        const _Float16* a2 = sMean + (size_t)(wv * 16 + row) * 64 + hsel * 32;
        float s = 0.f;
#pragma unroll
        for (int k = 0; k < 32; ++k) s += (float)a2[k] * sW3[hsel * 32 + k];
        float o = __shfl_xor(s, 16, 32);
        if (hsel == 0) {
            size_t g = g0 + (size_t)(wv * 16 + row);
            if (g < (size_t)G) bq[g] = s + o + hb2[0];
        }
    }
}

// ---------------------------------------------------------------------------
// Kernel 4: q_alt = Lin2(ELU(Lin1(enc[p] + (sums[g]-key1[p])*inv)))  -> [M,2]
// ---------------------------------------------------------------------------
__global__ __launch_bounds__(256)
void qalt_kernel(const float* __restrict__ enc, const float* __restrict__ key1,
                 const float* __restrict__ sums, const float* __restrict__ cnts,
                 const int* __restrict__ mp, const int* __restrict__ mg,
                 const float* __restrict__ w1, const float* __restrict__ b1,
                 const float* __restrict__ w2, const float* __restrict__ b2,
                 float* __restrict__ qout, int M, int G)
{
    __shared__ _Float16 sW1[64 * 64];   // [n][k]
    __shared__ float    sB1[64];
    __shared__ float    sW2f[2 * 64];   // [o][k]
    __shared__ float    sB2[2];
    __shared__ int      sPair[128];
    __shared__ int      sGrp[128];
    __shared__ float    sInv[128];
    __shared__ _Float16 sAlt[128 * 64];
    // overlay: fp32 async-gathered enc rows (phase 1-2) vs f16 hidden act (phase 3)
    __shared__ union {
        float    enc32[128 * 64];     // 32 KB
        _Float16 h1[128 * 64];        // 16 KB
    } sOv;

    const int t = threadIdx.x;
    for (int j = t; j < 64 * 64; j += 256) { int n = j >> 6, k = j & 63; sW1[j] = (_Float16)w1[k * 64 + n]; }
    if (t < 64)  sB1[t] = b1[t];
    if (t < 128) sW2f[t] = w2[(t & 63) * 2 + (t >> 6)];
    if (t < 2)   sB2[t] = b2[t];

    const size_t e0 = (size_t)blockIdx.x * 128;
    if (t < 128) {
        size_t e = e0 + t;
        int p = 0, g = 0; float inv = 0.f;
        if (e < (size_t)M) {
            p = mp[e]; g = mg[e];
            inv = 1.f / fmaxf(cnts[g], 1.f);
            __builtin_prefetch(key1 + (size_t)p * 64, 0, 0);   // global_prefetch_b8
        }
        sPair[t] = p; sGrp[t] = g; sInv[t] = inv;
    }
    __syncthreads();

    // async-gather the 128 enc rows (256B each) straight into LDS (ASYNCcnt path)
#pragma unroll
    for (int i = 0; i < 8; ++i) {
        int c = t + 256 * i;          // 16B chunk id, 0..2047
        int r = c >> 4;               // edge slot 0..127
        int off = (c & 15) * 4;       // float offset in row
        async_copy16(enc + (size_t)sPair[r] * 64 + off, sOv.enc32 + r * 64 + off);
    }
    async_wait0();
    __syncthreads();

    // build alt rows (thread t owns column k=t&63 -> coalesced row reads)
    for (int j = t; j < 128 * 64; j += 256) {
        int e = j >> 6, k = j & 63;
        int p = sPair[e], g = sGrp[e];
        float inv = sInv[e];
        float a = sOv.enc32[j] + (sums[(size_t)g * 64 + k] - key1[(size_t)p * 64 + k]) * inv;
        sAlt[j] = (_Float16)a;
    }
    __syncthreads();   // also separates sOv.enc32 reads from sOv.h1 writes

    const int lane = t & 31, wv = t >> 5;
    const int m = lane & 15;
    const int mhi = (lane & 16) ? 8 : 0;

    const _Float16* arow = sAlt + (size_t)(wv * 16 + m) * 64;
    v16h a0 = frag_a(arow, 0, lane);
    v16h a1 = frag_a(arow, 1, lane);
#pragma unroll
    for (int nt = 0; nt < 4; ++nt) {
        const int n = nt * 16 + (lane & 15);
        const float bv = sB1[n];
        v8f acc = { bv, bv, bv, bv, bv, bv, bv, bv };
        acc = WMMA_F16(a0, frag_b(sW1 + (size_t)n * 64, 0, lane), acc);
        acc = WMMA_F16(a1, frag_b(sW1 + (size_t)n * 64, 1, lane), acc);
#pragma unroll
        for (int r = 0; r < 8; ++r)
            sOv.h1[(size_t)(wv * 16 + r + mhi) * 64 + n] = (_Float16)eluf(acc[r]);
    }
    asm volatile("s_wait_dscnt 0x0" ::: "memory");

    // final: 64 -> 2; lane = (edge, output) pair
    {
        const int el = lane & 15, o = lane >> 4;
        const _Float16* h = sOv.h1 + (size_t)(wv * 16 + el) * 64;
        float s = sB2[o];
#pragma unroll
        for (int k = 0; k < 64; ++k) s += (float)h[k] * sW2f[o * 64 + k];
        size_t mi = e0 + (size_t)(wv * 16 + el);
        if (mi < (size_t)M) qout[mi * 2 + o] = s;
    }
}

// ---------------------------------------------------------------------------
extern "C" void kernel_launch(void* const* d_in, const int* in_sizes, int n_in,
                              void* d_out, int out_size, void* d_ws, size_t ws_size,
                              hipStream_t stream)
{
    const float* enc = (const float*)d_in[0];   // [P,64]
    const float* act = (const float*)d_in[1];   // [P,66]
    const int*   mp  = (const int*)d_in[2];     // [M]
    const int*   mg  = (const int*)d_in[3];     // [M]
    // d_in[4] = num_groups (device scalar; G derived from sizes instead)
    const float* w0 = (const float*)d_in[5];
    const float* w1 = (const float*)d_in[6];
    const float* w2 = (const float*)d_in[7];
    const float* w3 = (const float*)d_in[8];
    const float* w4 = (const float*)d_in[9];
    const float* w5 = (const float*)d_in[10];
    const float* b0 = (const float*)d_in[11];
    const float* b1 = (const float*)d_in[12];
    const float* b2 = (const float*)d_in[13];
    const float* b3 = (const float*)d_in[14];
    const float* b4 = (const float*)d_in[15];
    const float* b5 = (const float*)d_in[16];
    const float* hw0 = (const float*)d_in[17];
    const float* hw1 = (const float*)d_in[18];
    const float* hw2 = (const float*)d_in[19];
    const float* hb0 = (const float*)d_in[20];
    const float* hb1 = (const float*)d_in[21];
    const float* hb2 = (const float*)d_in[22];
    const float* l1w = (const float*)d_in[23];
    const float* l1b = (const float*)d_in[24];
    const float* l2w = (const float*)d_in[25];
    const float* l2b = (const float*)d_in[26];

    const int P = in_sizes[0] / 64;
    const int M = in_sizes[2];
    const int G = out_size - 2 * M;

    float* key1 = (float*)d_ws;                 // P*64
    float* sums = key1 + (size_t)P * 64;        // G*64
    float* cnts = sums + (size_t)G * 64;        // G
    (void)hipMemsetAsync(sums, 0, ((size_t)G * 64 + (size_t)G) * sizeof(float), stream);

    float* bq = (float*)d_out;   // [G]
    float* qa = bq + G;          // [M,2]

    mlp0_kernel<<<(P + 127) / 128, 256, 0, stream>>>(act, w0, w1, w2, w3, w4, w5,
                                                     b0, b1, b2, b3, b4, b5, key1, P);
    segsum_kernel<<<(M + 255) / 256, 256, 0, stream>>>(key1, mp, mg, sums, cnts, M, G);
    head_kernel<<<(G + 127) / 128, 256, 0, stream>>>(sums, cnts, hw0, hb0, hw1, hb1,
                                                     hw2, hb2, bq, G);
    qalt_kernel<<<(M + 127) / 128, 256, 0, stream>>>(enc, key1, sums, cnts, mp, mg,
                                                     l1w, l1b, l2w, l2b, qa, M, G);
}
